// CompleteObservationLearner_86388972192004
// MI455X (gfx1250) — matrix operand order repaired and checked by reading, not compiled, not run on hardware
//
#include <hip/hip_runtime.h>
#include <hip/hip_bf16.h>

// Problem sizes
#define E_DIM 4096
#define Q_DIM 16384
#define D_DIM 512

// ---- fused distance kernel tiling ----
#define TM 64        // E-panel rows per step
#define TN 128       // query columns per block
#define KCH 32       // k per WMMA
#define ESPLIT 4     // blocks split E into 4 slices (occupancy), merged later
#define BSTRIDE 520  // LDS row stride in bf16 elems (1040 B = 1024 B row + 16 B pad)

// ---- small GEMM (proj chain) tiling ----
#define GT 64
#define LPAD 40

// dynamic LDS layout for fused kernel (no static shared -> dynamic base at LDS offset 0)
#define OFF_B   0
#define SZ_B    (TN * BSTRIDE * 2)          // 133120
#define OFF_A0  (OFF_B + SZ_B)              // 133120
#define SZ_A    (TM * BSTRIDE * 2)          // 66560
#define OFF_A1  (OFF_A0 + SZ_A)             // 199680
#define OFF_PART (OFF_A1 + SZ_A)            // 266240
#define SMEM_BYTES (OFF_PART + 2 * TN * 12) // + pMin/pSum/pArg = 269312 (<= 320KB WGP LDS)

typedef __attribute__((ext_vector_type(16))) __bf16 v16bf;
typedef __attribute__((ext_vector_type(8)))  float  v8f;
typedef unsigned int u32x4 __attribute__((ext_vector_type(4)));
typedef int          i32x8 __attribute__((ext_vector_type(8)));
typedef int          i32x4 __attribute__((ext_vector_type(4)));

union FragU { v16bf v; uint4 u[2]; };

__device__ inline unsigned short f2bf(float x) {
    union { float f; unsigned u; } c; c.f = x;
    unsigned r = c.u + 0x7FFFu + ((c.u >> 16) & 1u);
    return (unsigned short)(r >> 16);
}
__device__ inline float bf2f(unsigned short h) {
    union { float f; unsigned u; } c; c.u = ((unsigned)h) << 16;
    return c.f;
}

// A fragment (16x32, MxK): lane -> M = row0+(l&15); elems 0..7 = K kb..kb+7, 8..15 = kb+16..23
__device__ inline v16bf fragA_ld(const unsigned short* base, int row, int stride, int k0, int lane) {
    FragU f;
    const unsigned short* p = base + row * stride + k0 + ((lane >> 4) * 8);
    f.u[0] = *(const uint4*)(p);
    f.u[1] = *(const uint4*)(p + 16);
    return f.v;
}
// B fragment (32x16, KxN) staged as Bt[n][k]: lane -> N = row0+(l&15); elems 0..7 = K kb..kb+7, 8..15 = kb+16..23
__device__ inline v16bf fragB_ld(const unsigned short* base, int row, int stride, int k0, int lane) {
    FragU f;
    const unsigned short* p = base + row * stride + k0 + ((lane >> 4) * 8);
    f.u[0] = *(const uint4*)(p);
    f.u[1] = *(const uint4*)(p + 16);
    return f.v;
}

// ---- Tensor Data Mover: 2D bf16 tile (512 x rows) -> LDS with 16B/row padding ----
// D# per CDNA5 ISA ch.8: group0 {count=1, lds_addr, global_addr, type=2},
// group1 {data_size=2B, pad_enable, pad_interval=256DW(1024B), pad_amount=4DW(16B),
//         tensor_dim0=512, tensor_dim1=rows, tile_dim0=512, tile_dim1=rows, dim0_stride=512}
// clang-23 toolchain: 6-arg builtin (g0 v4u, g1 v8i, g2 v4i, g3 v4i, extra v8i, cpol)
__device__ inline void tdm_load_panel(unsigned ldsByteOff, const void* gaddr, unsigned rows) {
    unsigned long long ga = (unsigned long long)gaddr;
    u32x4 g0;
    g0[0] = 1u;                                              // count=1 (valid descriptor)
    g0[1] = ldsByteOff;                                      // lds_addr (bytes)
    g0[2] = (unsigned)(ga & 0xFFFFFFFFu);                    // global_addr[31:0]
    g0[3] = (unsigned)((ga >> 32) & 0x01FFFFFFu) | (2u << 30); // addr[56:32] | type=2
    i32x8 g1;
    g1[0] = (1 << 16)    // data_size = 2 bytes
          | (1 << 20)    // pad_enable
          | (7 << 22)    // pad_interval: 256 DWORDs = 1024 B (one 512-elem bf16 row)
          | (3 << 25);   // pad_amount: 4 DWORDs = 16 B  -> LDS row stride 1040 B
    g1[1] = (int)((512u & 0xFFFFu) << 16);  // tensor_dim0[15:0]=512 (bits 79:64 are 0)
    g1[2] = (int)(rows << 16);              // tensor_dim0 hi=0 | tensor_dim1[15:0]=rows
    g1[3] = (int)(512u << 16);              // tensor_dim1 hi=0 | tile_dim0=512
    g1[4] = (int)rows;                      // tile_dim1=rows, tile_dim2=0
    g1[5] = 512;                            // tensor_dim0_stride lo32 = 512 elems
    g1[6] = 0;
    g1[7] = 0;
    i32x4 z4 = {0, 0, 0, 0};                // group2 / group3 unused (2D tile)
    i32x8 z8 = {0, 0, 0, 0, 0, 0, 0, 0};    // trailing group (gather/restore) unused
    __builtin_amdgcn_tensor_load_to_lds(g0, g1, z4, z4, z8, 0);
}

// ---- online softmax-min state (min of d2, argmin, sum exp(min - d2)) ----
struct OSM { float m; float s; int a; };

__device__ inline OSM osm_merge(OSM x, OSM y) {  // x covers earlier e than y (ties -> x)
    OSM r;
    if (y.m < x.m) { r.s = x.s * __expf(y.m - x.m) + y.s; r.m = y.m; r.a = y.a; }
    else           { r.s = x.s + y.s * __expf(x.m - y.m); r.m = x.m; r.a = x.a; }
    return r;
}
__device__ inline OSM osm_shfl16(OSM v) {
    OSM o;
    o.m = __shfl_xor(v.m, 16, 32);
    o.s = __shfl_xor(v.s, 16, 32);
    o.a = __shfl_xor(v.a, 16, 32);
    return o;
}
__device__ inline OSM tile_reduce(v8f c, int eBase, float qv, const float* __restrict__ pn) {
    OSM r; r.m = 3.0e38f; r.s = 0.0f; r.a = 0;
    #pragma unroll
    for (int i = 0; i < 8; ++i) {
        float d2 = fmaxf(pn[eBase + i] + qv - 2.0f * c[i], 0.0f);
        if (d2 < r.m) { r.s = r.s * __expf(d2 - r.m) + 1.0f; r.m = d2; r.a = eBase + i; }
        else          { r.s += __expf(r.m - d2); }
    }
    return r;
}

// ---------------- conversion / norm kernels ----------------

__global__ void cvt_bf16(const float* __restrict__ in, unsigned short* __restrict__ out, int n) {
    int i = blockIdx.x * blockDim.x + threadIdx.x;
    int stride = gridDim.x * blockDim.x;
    for (; i < n; i += stride) out[i] = f2bf(in[i]);
}

__global__ void cvt_w_t(const float* __restrict__ w, unsigned short* __restrict__ wt) {
    int i = blockIdx.x * blockDim.x + threadIdx.x;
    if (i >= D_DIM * D_DIM) return;
    int n = i >> 9, k = i & (D_DIM - 1);
    wt[i] = f2bf(w[k * D_DIM + n]);
}

__global__ void row_norm(const unsigned short* __restrict__ X, float* __restrict__ out, int rows) {
    int r = blockIdx.x * blockDim.x + threadIdx.x;
    if (r >= rows) return;
    const unsigned short* p = X + (size_t)r * D_DIM;
    float s = 0.0f;
    for (int d = 0; d < D_DIM; d += 8) {
        uint4 u = *(const uint4*)(p + d);
        const unsigned short* h = (const unsigned short*)&u;
        #pragma unroll
        for (int i = 0; i < 8; ++i) { float v = bf2f(h[i]); s += v * v; }
    }
    out[r] = s;
}

// ---------------- bf16 WMMA GEMM for the projection chain ----------------

__global__ __launch_bounds__(256) void gemm_bf16(const unsigned short* __restrict__ A,
                                                 const unsigned short* __restrict__ Bt,
                                                 const float* __restrict__ bias,
                                                 unsigned short* __restrict__ Out,
                                                 int relu) {
    __shared__ alignas(16) unsigned short lA[GT * LPAD];
    __shared__ alignas(16) unsigned short lB[GT * LPAD];
    const int t = threadIdx.x;
    const int lane = t & 31;
    const int wave = t >> 5;
    const int m0 = blockIdx.x * GT;
    const int n0 = blockIdx.y * GT;
    const int mT  = wave & 3;
    const int nT0 = (wave >> 2) * 2;
    const int lr = t >> 2;
    const int lc = (t & 3) * 8;

    v8f c0 = {}; v8f c1 = {};
    for (int k0 = 0; k0 < D_DIM; k0 += KCH) {
        *(uint4*)&lA[lr * LPAD + lc] = *(const uint4*)&A[(size_t)(m0 + lr) * D_DIM + k0 + lc];
        *(uint4*)&lB[lr * LPAD + lc] = *(const uint4*)&Bt[(size_t)(n0 + lr) * D_DIM + k0 + lc];
        __syncthreads();
        v16bf a  = fragA_ld(lA, mT * 16 + (lane & 15), LPAD, 0, lane);
        v16bf b0 = fragB_ld(lB, nT0 * 16 + (lane & 15), LPAD, 0, lane);
        v16bf b1 = fragB_ld(lB, (nT0 + 1) * 16 + (lane & 15), LPAD, 0, lane);
        c0 = __builtin_amdgcn_wmma_f32_16x16x32_bf16(false, a, false, b0, (short)0, c0, false, false);
        c1 = __builtin_amdgcn_wmma_f32_16x16x32_bf16(false, a, false, b1, (short)0, c1, false, false);
        __syncthreads();
    }
    const int nl = lane & 15;
    const int mb = (lane >> 4) * 8;
    #pragma unroll
    for (int j = 0; j < 2; ++j) {
        v8f c = j ? c1 : c0;
        int ncol = n0 + (nT0 + j) * 16 + nl;
        float bv = bias[ncol];
        #pragma unroll
        for (int i = 0; i < 8; ++i) {
            float v = c[i] + bv;
            if (relu) v = fmaxf(v, 0.0f);
            Out[(size_t)(m0 + mT * 16 + mb + i) * D_DIM + ncol] = f2bf(v);
        }
    }
}

// ---------------- fused distance GEMM + per-column argmin/logsumexp ----------------
// Block (bx,by): q-tile [bx*128, +128), E-slice [by*1024, +1024).
// TDM keeps B (query tile, 128x512) resident and double-buffers A (proj) panels.

__global__ __launch_bounds__(256) void fused_dist(const unsigned short* __restrict__ P,
                                                  const unsigned short* __restrict__ Qb,
                                                  const float* __restrict__ pn,
                                                  const float* __restrict__ qn,
                                                  float* __restrict__ pmin4,   // [ESPLIT][Q]
                                                  float* __restrict__ psum4,   // [ESPLIT][Q]
                                                  int*   __restrict__ parg4) { // [ESPLIT][Q]
    extern __shared__ char smem[];
    const unsigned short* lB = (const unsigned short*)(smem + OFF_B);
    float* pMin = (float*)(smem + OFF_PART);
    float* pSum = pMin + 2 * TN;
    int*   pArg = (int*)(pSum + 2 * TN);

    const int t = threadIdx.x;
    const int lane = t & 31;
    const int wave = t >> 5;
    const int q0 = blockIdx.x * TN;
    const int e_begin = blockIdx.y * (E_DIM / ESPLIT);
    const int nPanels = (E_DIM / ESPLIT) / TM;   // 16

    const int wm = wave & 1;        // 2 M wave-groups (each covers 2 subtiles = 32 rows)
    const int wn = wave >> 1;       // 4 N wave-groups (each covers 2 subtiles = 32 cols)
    const int nl = lane & 15;
    const int mb = (lane >> 4) * 8;

    // fixed query columns for this wave's two N-subtiles
    float qv0 = qn[q0 + (wn * 2 + 0) * 16 + nl];
    float qv1 = qn[q0 + (wn * 2 + 1) * 16 + nl];

    // kick off resident B load + first A panel via the Tensor Data Mover
    if (wave == 0) {
        tdm_load_panel(OFF_B, Qb + (size_t)q0 * D_DIM, TN);
        tdm_load_panel(OFF_A0, P + (size_t)e_begin * D_DIM, TM);
        __builtin_amdgcn_s_wait_tensorcnt(0);
    }
    __syncthreads();

    OSM run; run.m = 3.0e38f; run.s = 0.0f; run.a = 0;

    for (int p = 0; p < nPanels; ++p) {
        const int e0 = e_begin + p * TM;
        const unsigned short* curA =
            (const unsigned short*)(smem + ((p & 1) ? OFF_A1 : OFF_A0));

        // prefetch next A panel into the other buffer (overlaps with all compute below)
        if (wave == 0 && p + 1 < nPanels)
            tdm_load_panel((p & 1) ? OFF_A0 : OFF_A1,
                           P + (size_t)(e0 + TM) * D_DIM, TM);

        v8f a00 = {}, a01 = {}, a10 = {}, a11 = {};
        #pragma unroll 4
        for (int kc = 0; kc < D_DIM / KCH; ++kc) {
            const int k0 = kc * KCH;
            v16bf fa0 = fragA_ld(curA, (wm * 2 + 0) * 16 + nl, BSTRIDE, k0, lane);
            v16bf fa1 = fragA_ld(curA, (wm * 2 + 1) * 16 + nl, BSTRIDE, k0, lane);
            v16bf fb0 = fragB_ld(lB, (wn * 2 + 0) * 16 + nl, BSTRIDE, k0, lane);
            v16bf fb1 = fragB_ld(lB, (wn * 2 + 1) * 16 + nl, BSTRIDE, k0, lane);
            a00 = __builtin_amdgcn_wmma_f32_16x16x32_bf16(false, fa0, false, fb0, (short)0, a00, false, false);
            a01 = __builtin_amdgcn_wmma_f32_16x16x32_bf16(false, fa0, false, fb1, (short)0, a01, false, false);
            a10 = __builtin_amdgcn_wmma_f32_16x16x32_bf16(false, fa1, false, fb0, (short)0, a10, false, false);
            a11 = __builtin_amdgcn_wmma_f32_16x16x32_bf16(false, fa1, false, fb1, (short)0, a11, false, false);
        }

        // in-register column reductions (C layout: lane holds 8 consecutive M of one column)
        #pragma unroll
        for (int ns = 0; ns < 2; ++ns) {
            float qv = ns ? qv1 : qv0;
            OSM t0 = tile_reduce(ns ? a01 : a00, e0 + (wm * 2 + 0) * 16 + mb, qv, pn);
            OSM t1 = tile_reduce(ns ? a11 : a10, e0 + (wm * 2 + 1) * 16 + mb, qv, pn);
            OSM u0 = osm_merge(t0, osm_shfl16(t0));   // rows 0-7 + 8-15 of subtile
            OSM u1 = osm_merge(t1, osm_shfl16(t1));
            OSM u  = osm_merge(u0, u1);               // full 32-row wave strip
            if (lane < 16) {
                int c = (wn * 2 + ns) * 16 + lane;
                pMin[wm * TN + c] = u.m;
                pSum[wm * TN + c] = u.s;
                pArg[wm * TN + c] = u.a;
            }
        }
        __syncthreads();
        if (t < TN) {   // column owner merges the 2 M wave-groups, then folds into running state
            OSM s0; s0.m = pMin[t];      s0.s = pSum[t];      s0.a = pArg[t];
            OSM s1; s1.m = pMin[TN + t]; s1.s = pSum[TN + t]; s1.a = pArg[TN + t];
            run = osm_merge(run, osm_merge(s0, s1));
        }
        // wait for prefetched panel, release partial buffers for next panel
        if (wave == 0 && p + 1 < nPanels) __builtin_amdgcn_s_wait_tensorcnt(0);
        __syncthreads();
    }

    if (t < TN) {
        int q = q0 + t;
        size_t o = (size_t)blockIdx.y * Q_DIM + q;
        pmin4[o] = run.m;
        psum4[o] = run.s;
        parg4[o] = run.a;
    }
}

// ---------------- d2[target[q], q] via direct dot products (one wave per query) ----------------

__global__ __launch_bounds__(256) void target_dist(const unsigned short* __restrict__ P,
                                                   const unsigned short* __restrict__ Qb,
                                                   const float* __restrict__ pn,
                                                   const float* __restrict__ qn,
                                                   const int*   __restrict__ targets,
                                                   float* __restrict__ tval) {
    int w = (blockIdx.x * blockDim.x + threadIdx.x) >> 5;  // global wave id == query id
    int lane = threadIdx.x & 31;
    if (w >= Q_DIM) return;
    int tq = targets[w];
    const unsigned short* pr = P + (size_t)tq * D_DIM + lane * 16;
    const unsigned short* qr = Qb + (size_t)w * D_DIM + lane * 16;
    float s = 0.0f;
    #pragma unroll
    for (int h = 0; h < 2; ++h) {
        uint4 up = *(const uint4*)(pr + h * 8);
        uint4 uq = *(const uint4*)(qr + h * 8);
        const unsigned short* hp = (const unsigned short*)&up;
        const unsigned short* hq = (const unsigned short*)&uq;
        #pragma unroll
        for (int i = 0; i < 8; ++i) s += bf2f(hp[i]) * bf2f(hq[i]);
    }
    for (int off = 16; off > 0; off >>= 1) s += __shfl_down(s, off, 32);
    if (lane == 0) tval[w] = fmaxf(pn[tq] + qn[w] - 2.0f * s, 0.0f);
}

// ---------------- merge ESPLIT partial softmax states; emit preds/loss terms ----------------

__global__ void merge_cols(const float* __restrict__ pmin4, const float* __restrict__ psum4,
                           const int* __restrict__ parg4, const float* __restrict__ tval,
                           const int* __restrict__ targets,
                           float* __restrict__ preds, float* __restrict__ lossTerm,
                           float* __restrict__ corr) {
    int q = blockIdx.x * blockDim.x + threadIdx.x;
    if (q >= Q_DIM) return;
    OSM r; r.m = 3.0e38f; r.s = 0.0f; r.a = 0;
    #pragma unroll
    for (int j = 0; j < ESPLIT; ++j) {
        OSM s; s.m = pmin4[(size_t)j * Q_DIM + q];
        s.s = psum4[(size_t)j * Q_DIM + q];
        s.a = parg4[(size_t)j * Q_DIM + q];
        r = osm_merge(r, s);
    }
    int tq = targets[q];
    preds[q] = (float)r.a;
    float lse = -r.m + __logf(r.s);        // logsumexp(-d2[:,q])
    lossTerm[q] = tval[q] + lse;           // -log_softmax(-d2)[target,q]
    corr[q] = (r.a == tq) ? 1.0f : 0.0f;
}

// ---------------- deterministic final reduction ----------------

__global__ __launch_bounds__(256) void final_reduce(const float* __restrict__ lossTerm,
                                                    const float* __restrict__ corr,
                                                    float* __restrict__ out2) {
    __shared__ float sL[256];
    __shared__ float sC[256];
    int t = threadIdx.x;
    float l = 0.0f, c = 0.0f;
    for (int q = t; q < Q_DIM; q += 256) { l += lossTerm[q]; c += corr[q]; }
    sL[t] = l; sC[t] = c; __syncthreads();
    for (int s = 128; s > 0; s >>= 1) {
        if (t < s) { sL[t] += sL[t + s]; sC[t] += sC[t + s]; }
        __syncthreads();
    }
    if (t == 0) { out2[0] = sL[0] / (float)Q_DIM; out2[1] = sC[0] / (float)Q_DIM; }
}

// ---------------- host launch ----------------

extern "C" void kernel_launch(void* const* d_in, const int* in_sizes, int n_in,
                              void* d_out, int out_size, void* d_ws, size_t ws_size,
                              hipStream_t stream) {
    const float* env = (const float*)d_in[0];
    const float* qry = (const float*)d_in[1];
    const int*   tgt = (const int*)d_in[2];
    const float* w1  = (const float*)d_in[3];
    const float* b1  = (const float*)d_in[4];
    const float* w2  = (const float*)d_in[5];
    const float* b2  = (const float*)d_in[6];
    float* out = (float*)d_out;   // [Q preds][loss][acc]

    char* ws = (char*)d_ws;
    size_t off = 0;
    auto alloc = [&](size_t bytes) -> char* {
        char* p = ws + off;
        off += (bytes + 255) & ~(size_t)255;
        return p;
    };
    unsigned short* qbf = (unsigned short*)alloc((size_t)Q_DIM * D_DIM * 2);
    unsigned short* ebf = (unsigned short*)alloc((size_t)E_DIM * D_DIM * 2);
    unsigned short* hbf = (unsigned short*)alloc((size_t)E_DIM * D_DIM * 2);
    unsigned short* pbf = (unsigned short*)alloc((size_t)E_DIM * D_DIM * 2);
    unsigned short* w1t = (unsigned short*)alloc((size_t)D_DIM * D_DIM * 2);
    unsigned short* w2t = (unsigned short*)alloc((size_t)D_DIM * D_DIM * 2);
    float* qn    = (float*)alloc((size_t)Q_DIM * 4);
    float* pn    = (float*)alloc((size_t)E_DIM * 4);
    float* pmin4 = (float*)alloc((size_t)ESPLIT * Q_DIM * 4);
    float* psum4 = (float*)alloc((size_t)ESPLIT * Q_DIM * 4);
    int*   parg4 = (int*)alloc((size_t)ESPLIT * Q_DIM * 4);
    float* tval  = (float*)alloc((size_t)Q_DIM * 4);
    float* lt    = (float*)alloc((size_t)Q_DIM * 4);
    float* cr    = (float*)alloc((size_t)Q_DIM * 4);
    (void)ws_size; (void)in_sizes; (void)n_in; (void)out_size;

    cvt_bf16<<<1024, 256, 0, stream>>>(qry, qbf, Q_DIM * D_DIM);
    cvt_bf16<<<1024, 256, 0, stream>>>(env, ebf, E_DIM * D_DIM);
    cvt_w_t<<<(D_DIM * D_DIM) / 256, 256, 0, stream>>>(w1, w1t);
    cvt_w_t<<<(D_DIM * D_DIM) / 256, 256, 0, stream>>>(w2, w2t);
    row_norm<<<Q_DIM / 256, 256, 0, stream>>>(qbf, qn, Q_DIM);

    dim3 gg(E_DIM / GT, D_DIM / GT);
    gemm_bf16<<<gg, 256, 0, stream>>>(ebf, w1t, b1, hbf, 1);  // h = relu(env@w1+b1)
    gemm_bf16<<<gg, 256, 0, stream>>>(hbf, w2t, b2, pbf, 0);  // proj = h@w2+b2
    row_norm<<<E_DIM / 256, 256, 0, stream>>>(pbf, pn, E_DIM);

    target_dist<<<(Q_DIM * 32) / 256, 256, 0, stream>>>(pbf, qbf, pn, qn, tgt, tval);

    fused_dist<<<dim3(Q_DIM / TN, ESPLIT), 256, SMEM_BYTES, stream>>>(
        pbf, qbf, pn, qn, pmin4, psum4, parg4);

    merge_cols<<<Q_DIM / 256, 256, 0, stream>>>(pmin4, psum4, parg4, tval, tgt, out, lt, cr);
    final_reduce<<<1, 256, 0, stream>>>(lt, cr, out + Q_DIM);
}
